// GraphEncoder_38517266711090
// MI455X (gfx1250) — compile-verified, matrix operand order checked by reading, not run on hardware
//
#include <hip/hip_runtime.h>
#include <hip/hip_bf16.h>

// ---------------- types ----------------
typedef __attribute__((ext_vector_type(16))) __bf16          vbf16x16;
typedef __attribute__((ext_vector_type(8)))  float           vfloat8;
typedef __attribute__((ext_vector_type(8)))  unsigned short  vus8;
typedef __attribute__((ext_vector_type(4)))  unsigned short  vus4;

union FragBF {
    vbf16x16 v;
    vus8     u[2];
};

__device__ __forceinline__ unsigned short f2bf(float f) {
    unsigned u = __builtin_bit_cast(unsigned, f);
    u += 0x7FFFu + ((u >> 16) & 1u);      // round-to-nearest-even
    return (unsigned short)(u >> 16);
}

// ---------------- degree / norm ----------------
__global__ void init_deg(float* deg, long long n) {
    long long i = (long long)blockIdx.x * blockDim.x + threadIdx.x;
    if (i < n) deg[i] = 1.0f;             // self-loop weight
}

__global__ void deg_accum(float* deg, const int* __restrict__ col,
                          const float* __restrict__ ew, long long e) {
    long long i = (long long)blockIdx.x * blockDim.x + threadIdx.x;
    if (i < e) atomicAdd(&deg[col[i]], ew[i]);
}

__global__ void compute_dinv(const float* __restrict__ deg, float* dinv, long long n) {
    long long i = (long long)blockIdx.x * blockDim.x + threadIdx.x;
    if (i < n) {
        float d = deg[i];
        dinv[i] = (d > 0.0f) ? rsqrtf(d) : 0.0f;
    }
}

__global__ void compute_norm(const float* __restrict__ dinv, const int* __restrict__ row,
                             const int* __restrict__ col, const float* __restrict__ ew,
                             float* nrm, long long e) {
    long long i = (long long)blockIdx.x * blockDim.x + threadIdx.x;
    if (i < e) nrm[i] = dinv[row[i]] * ew[i] * dinv[col[i]];
}

// ---------------- fp32 -> bf16 conversion (x4 vectorized) ----------------
__global__ void cvt_f32_bf16_vec4(const float* __restrict__ src, unsigned short* dst,
                                  long long n4) {
    long long i = (long long)blockIdx.x * blockDim.x + threadIdx.x;
    if (i >= n4) return;
    float4 v = ((const float4*)src)[i];
    vus4 h;
    h.x = f2bf(v.x); h.y = f2bf(v.y); h.z = f2bf(v.z); h.w = f2bf(v.w);
    ((vus4*)dst)[i] = h;
}

// stack W2 (64x256) and W3 (64x256) into one 128x256 bf16 weight
__global__ void pack_w23(const float* __restrict__ W2, const float* __restrict__ W3,
                         unsigned short* dst) {
    int idx = blockIdx.x * blockDim.x + threadIdx.x;
    if (idx >= 128 * 256) return;
    int f = idx >> 8, k = idx & 255;
    float v = (f < 64) ? W2[f * 256 + k] : W3[(f - 64) * 256 + k];
    dst[idx] = f2bf(v);
}

// ---------------- WMMA GEMM:  C[N x F] = A[N x K] * W[F x K]^T ----------------
// block tile 128x128, 8 waves (4 along M x 2 along N), wave tile 32x64,
// K staged in 32-wide bf16 LDS tiles (row stride padded to 40 ushorts).
// bf16 tiles are staged with GLOBAL_LOAD_ASYNC_TO_LDS_B128 (ASYNCcnt path).
#define BM 128
#define BN 128
#define BK 32
#define LDP 40

__device__ __forceinline__ void async_b128_to_lds(unsigned lds_byte_addr,
                                                  const void* gaddr) {
    asm volatile("global_load_async_to_lds_b128 %0, %1, off"
                 :: "v"(lds_byte_addr), "v"(gaddr)
                 : "memory");
}

template <bool ABF16>
__global__ __launch_bounds__(256)
void gemm_wmma(const void* __restrict__ Aptr, const unsigned short* __restrict__ Wbf,
               float* __restrict__ C, int N, int K, int F) {
    __shared__ unsigned short lA[BM][LDP];
    __shared__ unsigned short lB[BN][LDP];

    const int tid   = threadIdx.x;
    const int lane  = tid & 31;
    const int wid   = tid >> 5;
    const int waveM = wid & 3;     // 0..3  -> 32-row strip
    const int waveN = wid >> 2;    // 0..1  -> 64-col strip
    const int m0    = blockIdx.x * BM;
    const int n0    = blockIdx.y * BN;

    vfloat8 acc[2][4];
    const vfloat8 vzero = {0.f, 0.f, 0.f, 0.f, 0.f, 0.f, 0.f, 0.f};
#pragma unroll
    for (int i = 0; i < 2; i++)
#pragma unroll
        for (int j = 0; j < 4; j++) acc[i][j] = vzero;

    const int lm = lane & 15;
    const int kb = (lane < 16) ? 0 : 8;   // 16-bit A/B fragment K base per half-wave

    for (int k0 = 0; k0 < K; k0 += BK) {
        // ---- stage A tile (128x32 bf16) ----
        if (ABF16) {
            // async copy, no VGPR staging; OOB rows read mapped scratch and are
            // masked in the epilogue
            const unsigned short* A = (const unsigned short*)Aptr;
#pragma unroll
            for (int c = tid; c < BM * 4; c += 256) {       // 512 x 16B chunks
                int r = c >> 2, kq = c & 3;
                async_b128_to_lds((unsigned)(size_t)&lA[r][kq * 8],
                                  A + (size_t)(m0 + r) * K + k0 + kq * 8);
            }
        } else {
            // fp32 A, convert in-register (guarded)
            const float* A = (const float*)Aptr;
#pragma unroll
            for (int c = tid; c < BM * BK / 4; c += 256) {
                int r = c >> 3, kq = c & 7;
                int gr = m0 + r;
                float4 v = {0.f, 0.f, 0.f, 0.f};
                if (gr < N) v = *(const float4*)(A + (size_t)gr * K + k0 + kq * 4);
                vus4 h;
                h.x = f2bf(v.x); h.y = f2bf(v.y); h.z = f2bf(v.z); h.w = f2bf(v.w);
                *(vus4*)&lA[r][kq * 4] = h;
            }
        }
        // ---- stage W tile (128x32 bf16) async ----
#pragma unroll
        for (int c = tid; c < BN * 4; c += 256) {           // 512 x 16B chunks
            int r = c >> 2, kq = c & 3;
            async_b128_to_lds((unsigned)(size_t)&lB[r][kq * 8],
                              Wbf + (size_t)(n0 + r) * K + k0 + kq * 8);
        }
        asm volatile("s_wait_asynccnt 0" ::: "memory");
        __syncthreads();

        // ---- load fragments (ISA 16-bit A layout) ----
        FragBF a[2], b[4];
#pragma unroll
        for (int i = 0; i < 2; i++) {
            int r = waveM * 32 + i * 16 + lm;
            a[i].u[0] = *(const vus8*)&lA[r][kb];       // K 0-7  / 8-15
            a[i].u[1] = *(const vus8*)&lA[r][kb + 16];  // K 16-23 / 24-31
        }
#pragma unroll
        for (int j = 0; j < 4; j++) {
            int r = waveN * 64 + j * 16 + lm;
            b[j].u[0] = *(const vus8*)&lB[r][kb];
            b[j].u[1] = *(const vus8*)&lB[r][kb + 16];
        }
#pragma unroll
        for (int i = 0; i < 2; i++)
#pragma unroll
            for (int j = 0; j < 4; j++)
                acc[i][j] = __builtin_amdgcn_wmma_f32_16x16x32_bf16(
                    false, a[i].v, false, b[j].v, (short)0, acc[i][j], false, false);
        __syncthreads();
    }

    // ---- epilogue: f32 C layout (VGPR r8 -> M = r8 or r8+8) ----
    const int mhalf = (lane < 16) ? 0 : 8;
#pragma unroll
    for (int i = 0; i < 2; i++)
#pragma unroll
        for (int j = 0; j < 4; j++)
#pragma unroll
            for (int r8 = 0; r8 < 8; r8++) {
                int m = m0 + waveM * 32 + i * 16 + mhalf + r8;
                int n = n0 + waveN * 64 + j * 16 + lm;
                if (m < N) C[(size_t)m * F + n] = acc[i][j][r8];
            }
}

// ---------------- edge scatter:  out[col] += norm * t[row]  (float4 lanes) ----------------
__global__ void edge_scatter(const float* __restrict__ t, const int* __restrict__ row,
                             const int* __restrict__ col, const float* __restrict__ nrm,
                             float* out, long long total, int shiftF4) {
    long long idx = (long long)blockIdx.x * blockDim.x + threadIdx.x;
    if (idx >= total) return;
    long long e = idx >> shiftF4;
    int q = (int)(idx & ((1 << shiftF4) - 1));
    int r = row[e], c = col[e];
    float w = nrm[e];
    float4 v = ((const float4*)t)[((size_t)r << shiftF4) + q];
    float* o = out + ((((size_t)c << shiftF4) + q) << 2);
    atomicAdd(o + 0, v.x * w);
    atomicAdd(o + 1, v.y * w);
    atomicAdd(o + 2, v.z * w);
    atomicAdd(o + 3, v.w * w);
}

// ---------------- layer epilogues ----------------
// h = relu(out1 + dinv^2 * t1 + b1)  -> bf16
__global__ void finish1(const float* __restrict__ out1, const float* __restrict__ t1,
                        const float* __restrict__ dinv, const float* __restrict__ b1,
                        unsigned short* hbf, long long total) {
    long long idx = (long long)blockIdx.x * blockDim.x + threadIdx.x;
    if (idx >= total) return;
    long long i = idx >> 8;
    int f = (int)(idx & 255);
    float di = dinv[i];
    float v = out1[idx] + di * di * t1[idx] + b1[f];
    hbf[idx] = f2bf(fmaxf(v, 0.0f));
}

// split [N,128] into mean (cols 0-63, +b2) and var (cols 64-127, +b3)
__global__ void finish2(const float* __restrict__ out23, const float* __restrict__ t23,
                        const float* __restrict__ dinv, const float* __restrict__ b2,
                        const float* __restrict__ b3, float* __restrict__ dout,
                        long long total, long long N) {
    long long idx = (long long)blockIdx.x * blockDim.x + threadIdx.x;
    if (idx >= total) return;
    long long i = idx >> 7;
    int f = (int)(idx & 127);
    float di = dinv[i];
    float v = out23[idx] + di * di * t23[idx];
    if (f < 64)
        dout[i * 64 + f] = v + b2[f];
    else
        dout[N * 64 + i * 64 + (f - 64)] = v + b3[f - 64];
}

// ---------------- host ----------------
extern "C" void kernel_launch(void* const* d_in, const int* in_sizes, int n_in,
                              void* d_out, int out_size, void* d_ws, size_t ws_size,
                              hipStream_t stream) {
    const float* x  = (const float*)d_in[0];
    const int*   ei = (const int*)d_in[1];
    const float* ew = (const float*)d_in[2];
    const float* W1 = (const float*)d_in[3];
    const float* b1 = (const float*)d_in[4];
    const float* W2 = (const float*)d_in[5];
    const float* b2 = (const float*)d_in[6];
    const float* W3 = (const float*)d_in[7];
    const float* b3 = (const float*)d_in[8];
    float* out = (float*)d_out;

    const long long N = in_sizes[0] / 512;   // 100000
    const long long E = in_sizes[2];         // 3.2M
    const int* row = ei;
    const int* col = ei + E;

    auto al = [](size_t b) { return (b + 255) & ~(size_t)255; };
    const size_t base_need =
        al(N * 4) + al(N * 4) + al(E * 4) + al((size_t)256 * 512 * 2) +
        al((size_t)128 * 256 * 2) + al((size_t)N * 256 * 2) +
        al((size_t)N * 256 * 4) + al((size_t)N * 256 * 4);
    const size_t xbf_bytes = al((size_t)N * 512 * 2);
    const bool use_xbf = ws_size >= base_need + xbf_bytes;   // constant per run

    // carve order matters: async A-tile OOB reads must land in mapped regions
    // (xbf spills into hbf, hbf spills into t1)
    char* p = (char*)d_ws;
    auto carve = [&](size_t bytes) -> void* {
        void* r = (void*)p;
        p += (bytes + 255) & ~(size_t)255;
        return r;
    };
    float*          deg   = (float*)carve(N * 4);
    float*          dinv  = (float*)carve(N * 4);
    float*          nrm   = (float*)carve(E * 4);
    unsigned short* w1bf  = (unsigned short*)carve((size_t)256 * 512 * 2);
    unsigned short* w23bf = (unsigned short*)carve((size_t)128 * 256 * 2);
    unsigned short* xbf   = use_xbf ? (unsigned short*)carve((size_t)N * 512 * 2) : nullptr;
    unsigned short* hbf   = (unsigned short*)carve((size_t)N * 256 * 2);
    float*          t1    = (float*)carve((size_t)N * 256 * 4);
    float*          big   = (float*)carve((size_t)N * 256 * 4);  // out1, later t23|out23
    float* t23   = big;
    float* out23 = big + N * 128;

    auto blk = [](long long n) { return (unsigned)((n + 255) / 256); };

    // 1) degrees + symmetric norm
    init_deg<<<blk(N), 256, 0, stream>>>(deg, N);
    deg_accum<<<blk(E), 256, 0, stream>>>(deg, col, ew, E);
    compute_dinv<<<blk(N), 256, 0, stream>>>(deg, dinv, N);
    compute_norm<<<blk(E), 256, 0, stream>>>(dinv, row, col, ew, nrm, E);

    // 2) weights -> bf16 (and optionally x -> bf16 for the async-A GEMM path)
    cvt_f32_bf16_vec4<<<blk(256 * 512 / 4), 256, 0, stream>>>(W1, w1bf, 256 * 512 / 4);
    pack_w23<<<blk(128 * 256), 256, 0, stream>>>(W2, W3, w23bf);

    // 3) layer 1: t1 = x * W1^T   (bf16 WMMA, fp32 accum)
    if (use_xbf) {
        cvt_f32_bf16_vec4<<<blk(N * 128), 256, 0, stream>>>(x, xbf, N * 128);
        gemm_wmma<true><<<dim3((unsigned)((N + 127) / 128), 2), 256, 0, stream>>>(
            (const void*)xbf, w1bf, t1, (int)N, 512, 256);
    } else {
        gemm_wmma<false><<<dim3((unsigned)((N + 127) / 128), 2), 256, 0, stream>>>(
            (const void*)x, w1bf, t1, (int)N, 512, 256);
    }

    // 4) propagate at 256 dims (L2-resident tables), then bias+ReLU -> bf16
    hipMemsetAsync(big, 0, (size_t)N * 256 * 4, stream);
    edge_scatter<<<blk(E * 64), 256, 0, stream>>>(t1, row, col, nrm, big, E * 64, 6);
    finish1<<<blk(N * 256), 256, 0, stream>>>(big, t1, dinv, b1, hbf, N * 256);

    // 5) fused layers 2+3: t23 = h * [W2;W3]^T  (bf16 WMMA, async-A)
    gemm_wmma<true><<<dim3((unsigned)((N + 127) / 128), 1), 256, 0, stream>>>(
        (const void*)hbf, w23bf, t23, (int)N, 256, 128);

    // 6) single 128-wide propagation, then split into mean/var with biases
    hipMemsetAsync(out23, 0, (size_t)N * 128 * 4, stream);
    edge_scatter<<<blk(E * 32), 256, 0, stream>>>(t23, row, col, nrm, out23, E * 32, 5);
    finish2<<<blk(N * 128), 256, 0, stream>>>(out23, t23, dinv, b2, b3, out, N * 128, N);
}
